// MoELookforward_38886633898788
// MI455X (gfx1250) — compile-verified, matrix operand order checked by reading, not run on hardware
//
#include <hip/hip_runtime.h>
#include <hip/hip_bf16.h>

// ---------------- problem constants ----------------
#define N_EMBD   1024
#define H4       4096      // 4*N_EMBD
#define N_EXPERT 8
#define TOPK     2
#define N_PRED   4
#define BATCH    2
#define SEQ      2048
#define T_TOK    4096      // BATCH*SEQ

typedef __attribute__((ext_vector_type(16))) __bf16 v16bf;
typedef __attribute__((ext_vector_type(8)))  float  v8f;

// d_out layout: [out T*H][router T*E][prior T*E][posterior T*E]
#define OUT_ROUTER ((size_t)T_TOK * N_EMBD)
#define OUT_PRIOR  (OUT_ROUTER + (size_t)T_TOK * N_EXPERT)
#define OUT_POST   (OUT_PRIOR  + (size_t)T_TOK * N_EXPERT)

// ---------------- 1) gating logits ----------------
__global__ __launch_bounds__(64) void moe_gate_kernel(
    const float* __restrict__ x, const float* __restrict__ Wp,
    const float* __restrict__ Wq, float* __restrict__ out) {
  const int t   = blockIdx.x;
  const int tid = threadIdx.x;
  const int b = t / SEQ, s = t % SEQ;
  const int sf = (s + N_PRED < SEQ) ? (s + N_PRED) : (SEQ - 1);
  const int tf = b * SEQ + sf;

  float xv[16], xf[16];
#pragma unroll
  for (int i = 0; i < 16; ++i) {
    xv[i] = x[(size_t)t  * N_EMBD + tid + i * 64];
    xf[i] = x[(size_t)tf * N_EMBD + tid + i * 64];
  }
  __shared__ float red[2][2];
  for (int e = 0; e < N_EXPERT; ++e) {
    float ap = 0.f, aq = 0.f;
#pragma unroll
    for (int i = 0; i < 16; ++i) {
      const float wp = Wp[e * N_EMBD + tid + i * 64];
      const float wq = Wq[e * N_EMBD + tid + i * 64];
      ap = fmaf(xv[i], wp, ap);
      aq = fmaf(xf[i], wq, aq);
    }
#pragma unroll
    for (int off = 16; off > 0; off >>= 1) {   // wave32 reduction
      ap += __shfl_down(ap, off);
      aq += __shfl_down(aq, off);
    }
    const int wave = tid >> 5;
    if ((tid & 31) == 0) { red[0][wave] = ap; red[1][wave] = aq; }
    __syncthreads();
    if (tid == 0) {
      const float p = red[0][0] + red[0][1];
      const float q = red[1][0] + red[1][1];
      out[OUT_PRIOR  + (size_t)t * N_EXPERT + e] = p;
      out[OUT_ROUTER + (size_t)t * N_EXPERT + e] = q;  // router == posterior
      out[OUT_POST   + (size_t)t * N_EXPERT + e] = q;
    }
    __syncthreads();
  }
}

// ---------------- 2) softmax + top-2 ----------------
__global__ void moe_route_kernel(const float* __restrict__ router,
                                 int* __restrict__ top_e, float* __restrict__ top_w) {
  const int t = blockIdx.x * blockDim.x + threadIdx.x;
  if (t >= T_TOK) return;
  float lg[N_EXPERT], p[N_EXPERT];
  float mx = -1e30f;
#pragma unroll
  for (int e = 0; e < N_EXPERT; ++e) { lg[e] = router[(size_t)t * N_EXPERT + e]; mx = fmaxf(mx, lg[e]); }
#pragma unroll
  for (int e = 0; e < N_EXPERT; ++e) p[e] = __expf(lg[e] - mx);
  int i0 = 0; float p0 = p[0];
#pragma unroll
  for (int e = 1; e < N_EXPERT; ++e) if (p[e] > p0) { p0 = p[e]; i0 = e; }
  int i1 = -1; float p1 = -1.f;
#pragma unroll
  for (int e = 0; e < N_EXPERT; ++e) if (e != i0 && p[e] > p1) { p1 = p[e]; i1 = e; }
  const float inv = 1.f / (p0 + p1);        // softmax denominator cancels
  top_e[2 * t + 0] = i0; top_e[2 * t + 1] = i1;
  top_w[2 * t + 0] = p0 * inv; top_w[2 * t + 1] = p1 * inv;
}

// ---------------- 3) f32 -> bf16 convert ----------------
__global__ void moe_cvt_kernel(const float* __restrict__ src, __bf16* __restrict__ dst, int n) {
  const int i = blockIdx.x * blockDim.x + threadIdx.x;
  if (i < n) dst[i] = (__bf16)src[i];
}

// ---------------- 4) deterministic per-expert dispatch ----------------
__global__ __launch_bounds__(256) void moe_dispatch_kernel(
    const int* __restrict__ top_e, int* __restrict__ tok,
    int* __restrict__ slot, int* __restrict__ counts) {
  const int e = blockIdx.x;
  const int tid = threadIdx.x;
  __shared__ int sd[256];
  int base = 0;
  for (int c = 0; c < T_TOK; c += 256) {
    const int t = c + tid;
    const int e0 = top_e[2 * t], e1 = top_e[2 * t + 1];
    const int k = (e0 == e) ? 0 : ((e1 == e) ? 1 : -1);
    const int flag = (k >= 0) ? 1 : 0;
    sd[tid] = flag;
    __syncthreads();
    for (int off = 1; off < 256; off <<= 1) {     // inclusive scan
      const int v = (tid >= off) ? sd[tid - off] : 0;
      __syncthreads();
      sd[tid] += v;
      __syncthreads();
    }
    if (flag) {
      const int pos = base + sd[tid] - 1;
      tok [e * T_TOK + pos] = t;
      slot[e * T_TOK + pos] = 2 * t + k;
    }
    base += sd[255];
    __syncthreads();
  }
  if (tid == 0) counts[e] = base;
}

// ------- WMMA fragment loader (ISA 16-bit A/B 16x32 per-lane layout) --------
// lane = half*16 + m ; elements 0..7 = K[half*8 .. +7], 8..15 = K[16+half*8 .. +7]
__device__ __forceinline__ v16bf load_frag_bf16(const __bf16* __restrict__ row,
                                                int kbase, int half) {
  const __bf16* p = row + kbase + half * 8;
  v16bf f;
#pragma unroll
  for (int i = 0; i < 8; ++i) { f[i] = p[i]; f[i + 8] = p[i + 16]; }
  return f;
}

// ------- async stage of a 128(cols) x 32(k) bf16 B-tile into LDS ------------
// gcol0 = first column row-pointer base; ldk = elements per weight row.
// 512 x 16B chunks, 2 per thread, via global_load_async_to_lds_b128 (ASYNCcnt)
__device__ __forceinline__ void stage_B_async(const __bf16* __restrict__ gcol0,
                                              size_t ldk, int kstep,
                                              __bf16* sbuf, int tid) {
#pragma unroll
  for (int j = 0; j < 2; ++j) {
    const int c   = tid + j * 256;
    const int col = c >> 2, part = c & 3;
    const __bf16* g = gcol0 + (size_t)col * ldk + kstep + part * 8;
    __bf16* l = sbuf + col * 32 + part * 8;
    const unsigned lofs = (unsigned)(uintptr_t)(void*)l;  // LDS addr = flat[31:0]
    asm volatile("global_load_async_to_lds_b128 %0, %1, off"
                 :: "v"(lofs), "v"(g) : "memory");
  }
}
__device__ __forceinline__ void wait_async_all() {
  asm volatile("s_wait_asynccnt 0x0" ::: "memory");
}

// ---------------- 5) GEMM1: act = relu(x @ W1b[e]^T)^2  (bf16 out) ----------
// block = 256 thr = 8 waves (4 M-waves x 2 N-waves), tile 128x128, K-step 32
__global__ __launch_bounds__(256) void moe_gemm1_kernel(
    const __bf16* __restrict__ xb, const __bf16* __restrict__ W1b,
    const int* __restrict__ tok, const int* __restrict__ slot,
    const int* __restrict__ counts, __bf16* __restrict__ act) {
  const int e   = blockIdx.z;
  const int cnt = counts[e];
  const int mbase = blockIdx.y * 128;
  if (mbase >= cnt) return;                        // block-uniform exit
  const int nbase = blockIdx.x * 128;
  const int tid  = threadIdx.x;
  const int lane = tid & 31;
  const int wave = tid >> 5;
  const int wm = wave >> 1, wn = wave & 1;
  const int half = lane >> 4, lm = lane & 15;

  __shared__ __bf16 sB[2][128 * 32];               // 2 x 8KB double buffer

  const __bf16* arow[2];
#pragma unroll
  for (int tm = 0; tm < 2; ++tm) {
    int r = mbase + wm * 32 + tm * 16 + lm;
    if (r >= cnt) r = cnt - 1;                     // clamp gather (store guarded)
    arow[tm] = xb + (size_t)tok[e * T_TOK + r] * N_EMBD;
  }
  const __bf16* gcol0 = W1b + ((size_t)e * H4 + nbase) * N_EMBD;

  stage_B_async(gcol0, N_EMBD, 0, sB[0], tid);
  wait_async_all();
  __syncthreads();

  v8f acc[2][4] = {};
  int buf = 0;
  for (int k = 0; k < N_EMBD; k += 32) {
    if (k + 32 < N_EMBD) stage_B_async(gcol0, N_EMBD, k + 32, sB[buf ^ 1], tid);
    v16bf af[2], bfr[4];
#pragma unroll
    for (int tm = 0; tm < 2; ++tm) af[tm] = load_frag_bf16(arow[tm], k, half);
#pragma unroll
    for (int tn = 0; tn < 4; ++tn)
      bfr[tn] = load_frag_bf16(&sB[buf][(wn * 64 + tn * 16 + lm) * 32], 0, half);
#pragma unroll
    for (int tm = 0; tm < 2; ++tm)
#pragma unroll
      for (int tn = 0; tn < 4; ++tn)
        acc[tm][tn] = __builtin_amdgcn_wmma_f32_16x16x32_bf16(
            false, af[tm], false, bfr[tn], (short)0, acc[tm][tn], false, false);
    if (k + 32 < N_EMBD) wait_async_all();
    __syncthreads();
    buf ^= 1;
  }

  // C layout: VGPR r -> M = r + 8*half ; N = lane&15
#pragma unroll
  for (int tm = 0; tm < 2; ++tm) {
#pragma unroll
    for (int r = 0; r < 8; ++r) {
      const int mrow = mbase + wm * 32 + tm * 16 + r + half * 8;
      if (mrow < cnt) {
        const int s = slot[e * T_TOK + mrow];
        __bf16* orow = act + (size_t)s * H4 + nbase + wn * 64;
#pragma unroll
        for (int tn = 0; tn < 4; ++tn) {
          float v = acc[tm][tn][r];
          v = (v > 0.f) ? v * v : 0.f;             // relu(x)^2
          orow[tn * 16 + lm] = (__bf16)v;
        }
      }
    }
  }
}

// ---------------- 6) GEMM2: y2 = act @ W2b[e]^T  (f32 out, per-slot rows) ---
__global__ __launch_bounds__(256) void moe_gemm2_kernel(
    const __bf16* __restrict__ act, const __bf16* __restrict__ W2b,
    const int* __restrict__ slot, const int* __restrict__ counts,
    float* __restrict__ y2) {
  const int e   = blockIdx.z;
  const int cnt = counts[e];
  const int mbase = blockIdx.y * 128;
  if (mbase >= cnt) return;
  const int nbase = blockIdx.x * 128;
  const int tid  = threadIdx.x;
  const int lane = tid & 31;
  const int wave = tid >> 5;
  const int wm = wave >> 1, wn = wave & 1;
  const int half = lane >> 4, lm = lane & 15;

  __shared__ __bf16 sB[2][128 * 32];

  const __bf16* arow[2];
#pragma unroll
  for (int tm = 0; tm < 2; ++tm) {
    int r = mbase + wm * 32 + tm * 16 + lm;
    if (r >= cnt) r = cnt - 1;
    arow[tm] = act + (size_t)slot[e * T_TOK + r] * H4;
  }
  const __bf16* gcol0 = W2b + ((size_t)e * N_EMBD + nbase) * H4;

  stage_B_async(gcol0, H4, 0, sB[0], tid);
  wait_async_all();
  __syncthreads();

  v8f acc[2][4] = {};
  int buf = 0;
  for (int k = 0; k < H4; k += 32) {
    if (k + 32 < H4) stage_B_async(gcol0, H4, k + 32, sB[buf ^ 1], tid);
    v16bf af[2], bfr[4];
#pragma unroll
    for (int tm = 0; tm < 2; ++tm) af[tm] = load_frag_bf16(arow[tm], k, half);
#pragma unroll
    for (int tn = 0; tn < 4; ++tn)
      bfr[tn] = load_frag_bf16(&sB[buf][(wn * 64 + tn * 16 + lm) * 32], 0, half);
#pragma unroll
    for (int tm = 0; tm < 2; ++tm)
#pragma unroll
      for (int tn = 0; tn < 4; ++tn)
        acc[tm][tn] = __builtin_amdgcn_wmma_f32_16x16x32_bf16(
            false, af[tm], false, bfr[tn], (short)0, acc[tm][tn], false, false);
    if (k + 32 < H4) wait_async_all();
    __syncthreads();
    buf ^= 1;
  }

#pragma unroll
  for (int tm = 0; tm < 2; ++tm) {
#pragma unroll
    for (int r = 0; r < 8; ++r) {
      const int mrow = mbase + wm * 32 + tm * 16 + r + half * 8;
      if (mrow < cnt) {
        const int s = slot[e * T_TOK + mrow];
        float* orow = y2 + (size_t)s * N_EMBD + nbase + wn * 64;
#pragma unroll
        for (int tn = 0; tn < 4; ++tn) orow[tn * 16 + lm] = acc[tm][tn][r];
      }
    }
  }
}

// ---------------- 7) weighted combine ----------------
__global__ void moe_combine_kernel(const float* __restrict__ y2,
                                   const float* __restrict__ top_w,
                                   float* __restrict__ out) {
  const int i = blockIdx.x * blockDim.x + threadIdx.x;
  if (i >= T_TOK * N_EMBD) return;
  const int t = i >> 10;           // / N_EMBD
  const int h = i & (N_EMBD - 1);
  out[i] = top_w[2 * t + 0] * y2[(size_t)(2 * t + 0) * N_EMBD + h]
         + top_w[2 * t + 1] * y2[(size_t)(2 * t + 1) * N_EMBD + h];
}

// ---------------- host launcher ----------------
extern "C" void kernel_launch(void* const* d_in, const int* in_sizes, int n_in,
                              void* d_out, int out_size, void* d_ws, size_t ws_size,
                              hipStream_t stream) {
  const float* x  = (const float*)d_in[0];
  const float* Wp = (const float*)d_in[1];
  const float* Wq = (const float*)d_in[2];
  const float* W1 = (const float*)d_in[3];
  const float* W2 = (const float*)d_in[4];
  float* out = (float*)d_out;

  // workspace carve-up (~243 MB)
  __bf16* xb   = (__bf16*)d_ws;                                    // T*H bf16      (8 MB)
  __bf16* W1b  = xb  + (size_t)T_TOK * N_EMBD;                     // E*4H*H bf16   (64 MB)
  __bf16* W2b  = W1b + (size_t)N_EXPERT * H4 * N_EMBD;             // E*H*4H bf16   (64 MB)
  __bf16* act  = W2b + (size_t)N_EXPERT * N_EMBD * H4;             // 2T*4H bf16    (64 MB)
  float*  y2   = (float*)(act + (size_t)2 * T_TOK * H4);           // 2T*H f32      (32 MB)
  int*    tope = (int*)(y2 + (size_t)2 * T_TOK * N_EMBD);          // 2T int
  float*  topw = (float*)(tope + 2 * T_TOK);                       // 2T f32
  int*    tok  = (int*)(topw + 2 * T_TOK);                         // E*T int
  int*    slt  = tok + (size_t)N_EXPERT * T_TOK;                   // E*T int
  int*    cnts = slt + (size_t)N_EXPERT * T_TOK;                   // E int

  // 1) gating logits (fills router/prior/posterior outputs)
  moe_gate_kernel<<<T_TOK, 64, 0, stream>>>(x, Wp, Wq, out);
  // 2) softmax + top-2
  moe_route_kernel<<<T_TOK / 256, 256, 0, stream>>>(out + OUT_ROUTER, tope, topw);
  // 3) bf16 conversions: x, W1, W2
  const int nx = T_TOK * N_EMBD;
  const int nw = N_EXPERT * H4 * N_EMBD;
  moe_cvt_kernel<<<(nx + 255) / 256, 256, 0, stream>>>(x,  xb,  nx);
  moe_cvt_kernel<<<(nw + 255) / 256, 256, 0, stream>>>(W1, W1b, nw);
  moe_cvt_kernel<<<(nw + 255) / 256, 256, 0, stream>>>(W2, W2b, nw);
  // 4) deterministic dispatch lists
  moe_dispatch_kernel<<<N_EXPERT, 256, 0, stream>>>(tope, tok, slt, cnts);
  // 5) expert GEMM1 + relu^2 : grid (N=4096/128, Mmax=4096/128, E)
  moe_gemm1_kernel<<<dim3(H4 / 128, T_TOK / 128, N_EXPERT), 256, 0, stream>>>(
      xb, W1b, tok, slt, cnts, act);
  // 6) expert GEMM2 : grid (N=1024/128, Mmax, E)
  moe_gemm2_kernel<<<dim3(N_EMBD / 128, T_TOK / 128, N_EXPERT), 256, 0, stream>>>(
      act, W2b, slt, cnts, y2);
  // 7) combine top-2 rows
  moe_combine_kernel<<<(T_TOK * N_EMBD + 255) / 256, 256, 0, stream>>>(y2, topw, out);
}
